// VectorQuantizer_7705171329578
// MI455X (gfx1250) — matrix operand order repaired and checked by reading, not compiled, not run on hardware
//
#include <hip/hip_runtime.h>
#include <hip/hip_bf16.h>

// ---------------------------------------------------------------------------
// Problem constants (from reference)
// ---------------------------------------------------------------------------
#define BATCH     16
#define TLEN      2048
#define DDIM      512
#define NCODES    8192
#define BT        (BATCH * TLEN)          // 32768 rows
#define NTILES    (NCODES / 16)           // 512 code tiles of 16
#define ZQ_ELEMS  ((size_t)BT * DDIM)     // 16,777,216
#define MROWS     128                     // rows per workgroup (16 per wave x 8 waves)

typedef __attribute__((ext_vector_type(16))) __bf16 v16bf;
typedef __attribute__((ext_vector_type(8)))  float  v8f;

struct AShorts { unsigned short v[16]; };   // 32 bytes -> bit_cast to v16bf

__device__ __forceinline__ unsigned short f2bf(float f) {
    // round-to-nearest-even fp32 -> bf16
    unsigned u = __float_as_uint(f);
    unsigned r = (u + 0x7FFFu + ((u >> 16) & 1u)) >> 16;
    return (unsigned short)r;
}

// ---------------------------------------------------------------------------
// Kernel 1: codebook fp32 -> bf16 + per-code squared norms (fp32)
// grid 1024 x 256 threads; one wave per code
// ---------------------------------------------------------------------------
__global__ __launch_bounds__(256) void k_prep(const float* __restrict__ cb,
                                              unsigned short* __restrict__ cb_bf,
                                              float* __restrict__ cnorm) {
    int wave = threadIdx.x >> 5;
    int lane = threadIdx.x & 31;
    int code = blockIdx.x * 8 + wave;
    const float* src = cb + (size_t)code * DDIM;
    unsigned short* dst = cb_bf + (size_t)code * DDIM;
    float s = 0.0f;
    #pragma unroll
    for (int i = lane; i < DDIM; i += 32) {
        float v = src[i];
        s += v * v;
        dst[i] = f2bf(v);
    }
    #pragma unroll
    for (int m = 16; m; m >>= 1) s += __shfl_xor(s, m);
    if (lane == 0) cnorm[code] = s;
}

// ---------------------------------------------------------------------------
// Kernel 2: WMMA distance GEMM + argmin.
// One workgroup = 128 rows; each of the 8 waves owns a 16-row A tile
// (held entirely in 128 VGPRs as bf16 fragments) and sweeps ALL 512 code
// tiles in the same order -> B fragments are shared through the WGP$.
// dist(row, code) = ||e||^2 - 2 * <x, e>   (||x||^2 dropped: row-constant)
// ---------------------------------------------------------------------------
__global__ __launch_bounds__(256) void k_argmin(const float* __restrict__ z_e,
                                                const __bf16* __restrict__ cbk,
                                                const float* __restrict__ cnorm,
                                                int* __restrict__ ids_ws,
                                                float* __restrict__ out_ids) {
    const int t    = threadIdx.x;
    const int w    = t >> 5;
    const int lane = t & 31;
    const int m    = lane & 15;            // A row within tile / C column index
    const int hi   = lane >> 4;            // low/high half of wave
    const size_t rowbase = (size_t)blockIdx.x * MROWS;

    // ---- build the 16 A fragments (16x512 bf16) directly in registers ----
    // A fragments use contiguous K (same as B): lane<16 holds K {0..15}, lane>=16 {16..31}
    const float* zrow = z_e + (rowbase + 16 * w + m) * DDIM + hi * 16;
    v16bf afrag[16];
    #pragma unroll
    for (int kt = 0; kt < 16; ++kt) {
        AShorts as;
        #pragma unroll
        for (int i = 0; i < 8; ++i) as.v[i]     = f2bf(zrow[kt * 32 + i]);
        #pragma unroll
        for (int i = 0; i < 8; ++i) as.v[8 + i] = f2bf(zrow[kt * 32 + 8 + i]);
        afrag[kt] = __builtin_bit_cast(v16bf, as);
    }

    float minv[8];
    int   mini[8];
    #pragma unroll
    for (int j = 0; j < 8; ++j) { minv[j] = 3.402823e38f; mini[j] = 0; }

    for (int nt = 0; nt < NTILES; ++nt) {
        const int nb = nt << 4;
        // B fragment base: lane m -> code nb+m; halves split K 0..15 / 16..31
        const __bf16* bp = cbk + ((size_t)(nb + m) << 9) + (hi << 4);
        // warm the next code tile (speculative; dropped if OOB)
        __builtin_prefetch(bp + ((size_t)16 << 9), 0, 1);

        // two independent accumulation chains over K
        v8f acc0 = {0.f, 0.f, 0.f, 0.f, 0.f, 0.f, 0.f, 0.f};
        v8f acc1 = {0.f, 0.f, 0.f, 0.f, 0.f, 0.f, 0.f, 0.f};
        #pragma unroll
        for (int kt = 0; kt < 8; ++kt) {
            v16bf bf0 = *(const v16bf*)(bp + kt * 32);
            v16bf bf1 = *(const v16bf*)(bp + (kt + 8) * 32);
            acc0 = __builtin_amdgcn_wmma_f32_16x16x32_bf16(
                false, afrag[kt], false, bf0, (short)0, acc0, false, false);
            acc1 = __builtin_amdgcn_wmma_f32_16x16x32_bf16(
                false, afrag[kt + 8], false, bf1, (short)0, acc1, false, false);
        }

        const float cn = cnorm[nb + m];
        #pragma unroll
        for (int j = 0; j < 8; ++j) {
            float d = cn - 2.0f * (acc0[j] + acc1[j]);
            if (d < minv[j]) { minv[j] = d; mini[j] = nb + m; }
        }
    }

    // ---- reduce across the 16 columns (lanes within each half); low-index ties ----
    #pragma unroll
    for (int j = 0; j < 8; ++j) {
        #pragma unroll
        for (int mm = 1; mm < 16; mm <<= 1) {
            float ov = __shfl_xor(minv[j], mm);
            int   oi = __shfl_xor(mini[j], mm);
            if (ov < minv[j] || (ov == minv[j] && oi < mini[j])) {
                minv[j] = ov; mini[j] = oi;
            }
        }
    }
    // this wave covered ALL codes for its 16 rows -> write results directly
    if (m == 0) {
        #pragma unroll
        for (int j = 0; j < 8; ++j) {
            size_t row = rowbase + 16 * w + j + hi * 8;  // C rows 8..15 in high half
            ids_ws[row]  = mini[j];
            out_ids[row] = (float)mini[j];
        }
    }
}

// ---------------------------------------------------------------------------
// Kernel 3: gather z_q (fp32 codebook), write z_q_st, per-block loss partials
// grid 2048 x 256; block b handles rows [16b, 16b+16)
// ---------------------------------------------------------------------------
__global__ __launch_bounds__(256) void k_gather(const float* __restrict__ z_e,
                                                const float* __restrict__ cb,
                                                const int* __restrict__ ids,
                                                float* __restrict__ out_zq,
                                                float* __restrict__ partial) {
    __shared__ float red[8];
    const int t = threadIdx.x;
    const size_t base = (size_t)blockIdx.x * 16 * DDIM;
    float s = 0.0f;
    #pragma unroll
    for (int i = 0; i < 32; ++i) {
        size_t e  = base + t + 256 * i;
        int row   = (int)(e >> 9);
        int col   = (int)(e & 511);
        int id    = ids[row];
        float zq  = cb[(size_t)id * DDIM + col];
        float ze  = z_e[e];
        out_zq[e] = ze + (zq - ze);        // straight-through forward value
        float d   = zq - ze;
        s += d * d;
    }
    #pragma unroll
    for (int mm = 16; mm; mm >>= 1) s += __shfl_xor(s, mm);
    if ((t & 31) == 0) red[t >> 5] = s;
    __syncthreads();
    if (t == 0) {
        float tot = 0.0f;
        #pragma unroll
        for (int i = 0; i < 8; ++i) tot += red[i];
        partial[blockIdx.x] = tot;
    }
}

// ---------------------------------------------------------------------------
// Kernel 4: final deterministic reduction -> vq_loss = 1.25 * mean(sq)
// ---------------------------------------------------------------------------
__global__ __launch_bounds__(256) void k_final(const float* __restrict__ partial,
                                               float* __restrict__ out_loss) {
    __shared__ float red[256];
    float s = 0.0f;
    for (int i = threadIdx.x; i < 2048; i += 256) s += partial[i];
    red[threadIdx.x] = s;
    __syncthreads();
    for (int off = 128; off; off >>= 1) {
        if (threadIdx.x < off) red[threadIdx.x] += red[threadIdx.x + off];
        __syncthreads();
    }
    if (threadIdx.x == 0)
        *out_loss = red[0] * (1.25f / (float)ZQ_ELEMS);
}

// ---------------------------------------------------------------------------
// Launcher
// ---------------------------------------------------------------------------
extern "C" void kernel_launch(void* const* d_in, const int* in_sizes, int n_in,
                              void* d_out, int out_size, void* d_ws, size_t ws_size,
                              hipStream_t stream) {
    (void)in_sizes; (void)n_in; (void)out_size; (void)ws_size;

    const float* z_e = (const float*)d_in[0];   // [16, 2048, 512] fp32
    const float* cb  = (const float*)d_in[1];   // [8192, 512] fp32

    float* out      = (float*)d_out;
    float* out_zq   = out;                       // 16,777,216 floats
    float* out_ids  = out + ZQ_ELEMS;            // 32,768 floats
    float* out_loss = out + ZQ_ELEMS + BT;       // 1 float

    // workspace layout (~8.6 MB)
    char* ws = (char*)d_ws;
    unsigned short* cb_bf = (unsigned short*)ws;                    // 8,388,608 B
    float* cnorm   = (float*)(ws + (size_t)NCODES * DDIM * 2);      //   32,768 B
    int*   ids_ws  = (int*)  ((char*)cnorm + (size_t)NCODES * 4);   //  131,072 B
    float* partial = (float*)((char*)ids_ws + (size_t)BT * 4);      //    8,192 B

    k_prep  <<<NCODES / 8, 256, 0, stream>>>(cb, cb_bf, cnorm);
    k_argmin<<<BT / MROWS, 256, 0, stream>>>(z_e, (const __bf16*)cb_bf, cnorm,
                                             ids_ws, out_ids);
    k_gather<<<BT / 16,    256, 0, stream>>>(z_e, cb, ids_ws, out_zq, partial);
    k_final <<<1,          256, 0, stream>>>(partial, out_loss);
}
